// CentroidAwareInfoNCELoss_30812095381512
// MI455X (gfx1250) — matrix operand order repaired and checked by reading, not compiled, not run on hardware
//
#include <hip/hip_runtime.h>
#include <hip/hip_bf16.h>
#include <math.h>

// Problem constants (fixed by reference setup_inputs)
#define Bn   4
#define Cn   256
#define Hn   128
#define Wn   128
#define HW   (Hn*Wn)       // 16384
#define Np   (Bn*HW)       // 65536
#define KCLS 19
#define KPAD 32
#define IGN  255
#define MAXS 4096
#define GHg  512
#define INVT (1.0f/0.07f)

typedef __attribute__((ext_vector_type(16))) _Float16 v16h;
typedef __attribute__((ext_vector_type(8)))  float    v8f;

union AFrag { v16h v; _Float16 h[16]; };
union CFrag { v8f  v; float    f[8];  };

// ---------------- workspace layout (bytes) ----------------
// 0      : g_sums  19*256 f32  (19456)
// 19456  : g_cnt   19 f32
// 19532  : g_loss  1 f32          (sums/cnt/loss contiguous: zeroed together)
// 19584  : g_has   32 int
// 19712  : g_cent  32*256 f16  (32B aligned for v16h loads)
// 36096  : g_blkc  256 int
// 37120  : g_blko  256 int
// 38144  : g_S     1 int
// 38208  : g_sidx  4096 int
// 54592  : g_slab  4096 int
// 70976  : g_sinv  4096 f32    -> end 87360

__global__ void k_init(float* __restrict__ g) {
    int i = blockIdx.x * 256 + threadIdx.x;
    if (i < KCLS*Cn + KCLS + 1) g[i] = 0.0f;
}

// ---- Phase 1: per-class sums of L2-normalized target features ----
__global__ void k_centroid(const float* __restrict__ f_t,
                           const int*   __restrict__ tpl,
                           float* __restrict__ g_sums,
                           float* __restrict__ g_cnt) {
    __shared__ float s_sums[KCLS*Cn];
    __shared__ float s_cnt[KCLS];
    const int t = threadIdx.x;
    for (int i = t; i < KCLS*Cn; i += 256) s_sums[i] = 0.0f;
    if (t < KCLS) s_cnt[t] = 0.0f;
    __syncthreads();

    const int n  = blockIdx.x * 256 + t;
    const int b  = n >> 14;
    const int hw = n & (HW - 1);
    const int h  = hw >> 7;
    const int w  = hw & 127;
    const int lab = tpl[b*GHg*GHg + (h*4)*GHg + (w*4)];
    const int seg = (lab == IGN) ? KCLS : lab;
    const bool act = (seg >= 0) && (seg < KCLS);

    const float* base = f_t + (size_t)b * Cn * HW + hw;
    float ss = 0.0f;
    for (int c = 0; c < Cn; ++c) {
        if (c + 8 < Cn) __builtin_prefetch(&base[(size_t)(c+8) * HW], 0, 0);
        float v = base[(size_t)c * HW];
        ss += v * v;
    }
    const float inv = 1.0f / fmaxf(sqrtf(ss), 1e-12f);

    if (act) {
        atomicAdd(&s_cnt[seg], 1.0f);
        for (int c = 0; c < Cn; ++c) {
            float v = base[(size_t)c * HW] * inv;
            atomicAdd(&s_sums[seg*Cn + c], v);
        }
    }
    __syncthreads();
    for (int i = t; i < KCLS*Cn; i += 256) {
        float v = s_sums[i];
        if (v != 0.0f) atomicAdd(&g_sums[i], v);
    }
    if (t < KCLS) atomicAdd(&g_cnt[t], s_cnt[t]);
}

// ---- Phase 2: finalize centroids -> f16 table [32][256], zero-padded ----
__global__ void k_finalize(const float* __restrict__ g_sums,
                           const float* __restrict__ g_cnt,
                           int*   __restrict__ g_has,
                           _Float16* __restrict__ g_cent) {
    __shared__ float red[256];
    const int t = threadIdx.x;
    for (int k = 0; k < KPAD; ++k) {
        float cnt  = (k < KCLS) ? g_cnt[k] : 0.0f;
        float mval = (k < KCLS) ? g_sums[k*Cn + t] / fmaxf(cnt, 1.0f) : 0.0f;
        red[t] = mval * mval;
        __syncthreads();
        for (int off = 128; off > 0; off >>= 1) {
            if (t < off) red[t] += red[t + off];
            __syncthreads();
        }
        float invn = 1.0f / fmaxf(sqrtf(red[0]), 1e-12f);
        __syncthreads();
        g_cent[k*Cn + t] = (_Float16)(mval * invn);
        if (t == 0) g_has[k] = (k < KCLS && cnt > 0.0f) ? 1 : 0;
    }
}

__device__ __forceinline__ bool pix_valid(const int* __restrict__ sgt,
                                          const int* __restrict__ g_has,
                                          int n, int& labc) {
    const int b  = n >> 14;
    const int hw = n & (HW - 1);
    const int h  = hw >> 7;
    const int w  = hw & 127;
    const int lab = sgt[b*GHg*GHg + (h*4)*GHg + (w*4)];
    labc = lab < 0 ? 0 : (lab > KCLS-1 ? KCLS-1 : lab);
    return (lab != IGN) && (g_has[labc] != 0);
}

// ---- Phase 3: per-block valid counts (wave32 ballot) ----
__global__ void k_count(const int* __restrict__ sgt, const int* __restrict__ g_has,
                        int* __restrict__ g_blkc) {
    __shared__ int s_c[8];
    const int t = threadIdx.x;
    const int n = blockIdx.x * 256 + t;
    int labc;
    bool valid = pix_valid(sgt, g_has, n, labc);
    unsigned long long bal = __ballot(valid);
    const int lane = t & 31, wv = t >> 5;
    if (lane == 0) s_c[wv] = __popcll(bal);
    __syncthreads();
    if (t == 0) {
        int tot = 0;
        for (int i = 0; i < 8; ++i) tot += s_c[i];
        g_blkc[blockIdx.x] = tot;
    }
}

__global__ void k_scan(const int* __restrict__ g_blkc, int* __restrict__ g_blko,
                       int* __restrict__ g_S) {
    int acc = 0;
    for (int i = 0; i < Np/256; ++i) { g_blko[i] = acc; acc += g_blkc[i]; }
    g_S[0] = acc < MAXS ? acc : MAXS;
}

// ---- Phase 4: compact first-MAXS valid pixels (stable order) ----
__global__ void k_compact(const int* __restrict__ sgt, const int* __restrict__ g_has,
                          const int* __restrict__ g_blko,
                          int* __restrict__ sel_idx, int* __restrict__ sel_lab) {
    __shared__ int s_c[8];
    const int t = threadIdx.x;
    const int n = blockIdx.x * 256 + t;
    int labc;
    bool valid = pix_valid(sgt, g_has, n, labc);
    unsigned long long bal = __ballot(valid);
    const int lane = t & 31, wv = t >> 5;
    if (lane == 0) s_c[wv] = __popcll(bal);
    __syncthreads();
    int wbase = 0;
    for (int i = 0; i < wv; ++i) wbase += s_c[i];
    int lpre = __popcll(bal & ((1ull << lane) - 1ull));
    int rank = g_blko[blockIdx.x] + wbase + lpre;
    if (valid && rank < MAXS) { sel_idx[rank] = n; sel_lab[rank] = labc; }
}

// ---- Phase 5: inverse L2 norms of selected f_aug pixels ----
__global__ void k_selinv(const float* __restrict__ f_aug,
                         const int* __restrict__ sel_idx,
                         const int* __restrict__ gS,
                         float* __restrict__ sel_inv) {
    int s = blockIdx.x * 256 + threadIdx.x;
    if (s >= gS[0]) return;
    int n  = sel_idx[s];
    int b  = n >> 14;
    int hw = n & (HW - 1);
    const float* base = f_aug + (size_t)b * Cn * HW + hw;
    float ss = 0.0f;
    for (int c = 0; c < Cn; ++c) { float v = base[(size_t)c * HW]; ss += v * v; }
    sel_inv[s] = 1.0f / fmaxf(sqrtf(ss), 1e-12f);
}

// ---- Phase 6: WMMA sim GEMM + masked CE. 1 wave per 16-pixel tile ----
__global__ void k_ce(const float* __restrict__ f_aug,
                     const _Float16* __restrict__ cent,
                     const int* __restrict__ hascent,
                     const int* __restrict__ sel_idx,
                     const int* __restrict__ sel_lab,
                     const float* __restrict__ sel_inv,
                     const int* __restrict__ gS,
                     float* __restrict__ g_loss) {
    const int S = gS[0];
    const int tile = blockIdx.x;
    if (tile * 16 >= S) return;             // block-uniform: EXEC stays full for WMMA

    const int lane = threadIdx.x;           // wave32
    const int m    = lane & 15;
    const int half = lane >> 4;
    const int srow = tile * 16 + m;
    const bool rv  = (srow < S);
    const int   p   = rv ? sel_idx[srow] : sel_idx[0];
    const float inv = rv ? sel_inv[srow] : 0.0f;   // zero row for pad slots (no divergence)
    const int b  = p >> 14;
    const int hw = p & (HW - 1);
    const float* abase = f_aug + (size_t)b * Cn * HW + hw;

    v8f acc0 = {};   // classes 0..15
    v8f acc1 = {};   // classes 16..31 (19..31 are zero centroids)
    for (int ck = 0; ck < Cn; ck += 32) {
        // A-frag, 16x32, this code's contiguous K order (not the ISA table order): lane half -> K {h*16..h*16+15} (matches B)
        AFrag a;
#pragma unroll
        for (int i = 0; i < 8; ++i) {
            float v0 = abase[(size_t)(ck + half*16 +     i) * HW];
            float v1 = abase[(size_t)(ck + half*16 + 8 + i) * HW];
            a.h[i]     = (_Float16)(v0 * inv);
            a.h[8 + i] = (_Float16)(v1 * inv);
        }
        // B-frag: column-per-lane, 16 contiguous K values (32B aligned)
        v16h b0 = *(const v16h*)(cent + (size_t)m        * Cn + ck + half*16);
        v16h b1 = *(const v16h*)(cent + (size_t)(16 + m) * Cn + ck + half*16);
        acc0 = __builtin_amdgcn_wmma_f32_16x16x32_f16(false, a.v, false, b0,
                                                      (short)0, acc0, false, false);
        acc1 = __builtin_amdgcn_wmma_f32_16x16x32_f16(false, a.v, false, b1,
                                                      (short)0, acc1, false, false);
    }

    // C/D layout: lane<16 -> M=r,   N=lane;  lane>=16 -> M=8+r, N=lane-16
    __shared__ float s_sim[16][33];
    CFrag c0, c1; c0.v = acc0; c1.v = acc1;
#pragma unroll
    for (int r = 0; r < 8; ++r) {
        int row = half*8 + r;
        s_sim[row][m]      = c0.f[r] * INVT;
        s_sim[row][16 + m] = c1.f[r] * INVT;
    }
    __syncthreads();

    if (lane < 16) {
        int s2 = tile*16 + lane;
        if (s2 < S) {
            int lab = sel_lab[s2];
            float mx = -3.0e38f;
            for (int k = 0; k < KCLS; ++k)
                if (hascent[k]) mx = fmaxf(mx, s_sim[lane][k]);
            float se = 0.0f;
            for (int k = 0; k < KCLS; ++k)
                if (hascent[k]) se += expf(s_sim[lane][k] - mx);
            float ce = logf(se) + mx - s_sim[lane][lab];
            atomicAdd(g_loss, ce);
        }
    }
}

__global__ void k_final(const float* __restrict__ g_loss, const int* __restrict__ gS,
                        float* __restrict__ out) {
    out[0] = 1.0f * (g_loss[0] / fmaxf((float)gS[0], 1.0f));
}

extern "C" void kernel_launch(void* const* d_in, const int* in_sizes, int n_in,
                              void* d_out, int out_size, void* d_ws, size_t ws_size,
                              hipStream_t stream) {
    const float* f_aug = (const float*)d_in[0];
    const float* f_t   = (const float*)d_in[1];
    const int*   sgt   = (const int*)d_in[2];
    const int*   tpl   = (const int*)d_in[3];
    float* out = (float*)d_out;

    char* ws = (char*)d_ws;
    float*    g_sums = (float*)(ws + 0);
    float*    g_cnt  = (float*)(ws + 19456);
    float*    g_loss = (float*)(ws + 19532);
    int*      g_has  = (int*)  (ws + 19584);
    _Float16* g_cent = (_Float16*)(ws + 19712);
    int*      g_blkc = (int*)  (ws + 36096);
    int*      g_blko = (int*)  (ws + 37120);
    int*      g_S    = (int*)  (ws + 38144);
    int*      g_sidx = (int*)  (ws + 38208);
    int*      g_slab = (int*)  (ws + 54592);
    float*    g_sinv = (float*)(ws + 70976);

    k_init    <<<20, 256, 0, stream>>>(g_sums);
    k_centroid<<<Np/256, 256, 0, stream>>>(f_t, tpl, g_sums, g_cnt);
    k_finalize<<<1, 256, 0, stream>>>(g_sums, g_cnt, g_has, g_cent);
    k_count   <<<Np/256, 256, 0, stream>>>(sgt, g_has, g_blkc);
    k_scan    <<<1, 1, 0, stream>>>(g_blkc, g_blko, g_S);
    k_compact <<<Np/256, 256, 0, stream>>>(sgt, g_has, g_blko, g_sidx, g_slab);
    k_selinv  <<<MAXS/256, 256, 0, stream>>>(f_aug, g_sidx, g_S, g_sinv);
    k_ce      <<<MAXS/16, 32, 0, stream>>>(f_aug, g_cent, g_has, g_sidx, g_slab,
                                           g_sinv, g_S, g_loss);
    k_final   <<<1, 1, 0, stream>>>(g_loss, g_S, out);
}
